// DT_GPT_24361054502966
// MI455X (gfx1250) — matrix operand order repaired and checked by reading, not compiled, not run on hardware
//
#include <hip/hip_runtime.h>
#include <hip/hip_bf16.h>

// ---------------------------------------------------------------------------
// Decision-Transformer forward on gfx1250 (MI455X).
// Big GEMMs -> v_wmma_f32_16x16x32_bf16 (wave32), fp32 accumulate.
// Weights pre-swizzled to WMMA fragment order so every lane's B fragment is a
// single contiguous 32B load. Attention stages K/V via async global->LDS.
// ---------------------------------------------------------------------------

#define B_    128
#define T_    50
#define D_    768
#define NH_   12
#define HD_   64
#define NL_   6
#define V_    18
#define IL_   10
#define S3T   (3 * T_)          // 150
#define MTOK  (B_ * S3T)        // 19200
#define BT_   (B_ * T_)         // 6400
#define EPSLN 1e-5f

typedef __bf16 bf16;
typedef __attribute__((ext_vector_type(16))) __bf16 v16bf;
typedef __attribute__((ext_vector_type(8)))  __bf16 v8bf;
typedef __attribute__((ext_vector_type(8)))  float  v8f;
typedef __attribute__((ext_vector_type(4)))  int    v4i;

#if defined(__has_builtin)
#if __has_builtin(__builtin_amdgcn_global_load_async_to_lds_b128)
#define USE_ASYNC_LDS 1
#endif
#endif

__device__ __forceinline__ float bf2f(bf16 x) {
  unsigned short h = __builtin_bit_cast(unsigned short, x);
  unsigned u = ((unsigned)h) << 16;
  return __builtin_bit_cast(float, u);
}
__device__ __forceinline__ bf16 f2bf(float f) {
  unsigned u = __builtin_bit_cast(unsigned, f);
  unsigned lsb = (u >> 16) & 1u;
  u += 0x7FFFu + lsb;                       // round to nearest even
  unsigned short h = (unsigned short)(u >> 16);
  return __builtin_bit_cast(bf16, h);
}
__device__ __forceinline__ float sigmoidf_(float x) { return 1.0f / (1.0f + __expf(-x)); }

// ---------------------------------------------------------------------------
// Weight conversion f32 -> bf16 with WMMA-B fragment swizzle.
// Source W[K,N] row-major. Dest layout: tile (kb = k/32, nb = n/16) stores
// 32 lanes x 16 contiguous bf16: lane = (kr/16)*16 + (n%16), elem j = kr%16.
// GEMM then loads each lane's fragment as one 32-byte contiguous chunk.
// ---------------------------------------------------------------------------
__global__ void convert_swizzle_b_kernel(const float* __restrict__ W, bf16* __restrict__ out,
                                         int K, int N) {
  int idx = blockIdx.x * blockDim.x + threadIdx.x;
  if (idx >= K * N) return;
  int n = idx % N, k = idx / N;
  int kb = k >> 5, kr = k & 31;
  int lane = ((kr >> 4) << 4) + (n & 15);
  int j = kr & 15;
  size_t dst = ((size_t)(kb * (N >> 4) + (n >> 4)) * 32 + lane) * 16 + j;
  out[dst] = f2bf(W[idx]);
}

__global__ void zero_kernel(float* p, int n) {
  int i = blockIdx.x * blockDim.x + threadIdx.x;
  if (i < n) p[i] = 0.0f;
}

// ---------------------------------------------------------------------------
// WMMA GEMM: C[M,N] = act( A[M,K] @ B[K,N] + bias + resid )
// A row-major bf16; B pre-swizzled bf16 (see above); fp32 accumulate.
// Block = 8 waves (4x2); each wave owns a 16x32 output -> block tile 64x64.
// K unrolled x2 -> 4 v_wmma per loop body. M%64==0, N%64==0, K%64==0.
// ---------------------------------------------------------------------------
template <int ACT, bool RESID, bool OUTF, bool OUTBF>
__global__ void __launch_bounds__(256)
wmma_gemm_bf16(const bf16* __restrict__ A, const bf16* __restrict__ Bs,
               const float* __restrict__ bias, const float* resid,
               float* Cf, bf16* Cbf, int M, int N, int K) {
  const int lane = threadIdx.x & 31;
  const int wave = threadIdx.x >> 5;
  const int wy = wave >> 1;                   // 0..3
  const int wx = wave & 1;                    // 0..1
  const int m0 = blockIdx.x * 64 + wy * 16;
  const int n0 = blockIdx.y * 64 + wx * 32;   // wave covers two 16-wide n tiles
  const int half = lane >> 4;
  const int l15  = lane & 15;
  const int am   = m0 + l15;
  const int ntiles = N >> 4;
  const int nb0    = n0 >> 4;

  const bf16* Arow = A + (size_t)am * K;
  v8f acc0 = {}, acc1 = {};
  for (int k0 = 0; k0 < K; k0 += 64) {
    __builtin_prefetch(Arow + k0 + 64, 0, 1);           // global_prefetch_b8
#pragma unroll
    for (int u = 0; u < 2; ++u) {
      const int kk = k0 + u * 32;
      // A 16x32 fragment: lane's 16 values are one contiguous 32B span (k = 16*half + s).
      const bf16* Ap = Arow + kk + half * 16;
      v8bf a0 = *(const v8bf*)(Ap);
      v8bf a1 = *(const v8bf*)(Ap + 8);
      v16bf af = __builtin_shufflevector(a0, a1, 0, 1, 2, 3, 4, 5, 6, 7,
                                                 8, 9, 10, 11, 12, 13, 14, 15);
      // B fragments: one contiguous 32B chunk per lane per tile.
      const bf16* Bt = Bs + ((size_t)((kk >> 5) * ntiles + nb0) * 32 + lane) * 16;
      v16bf b0 = *(const v16bf*)(Bt);
      v16bf b1 = *(const v16bf*)(Bt + 512);
      acc0 = __builtin_amdgcn_wmma_f32_16x16x32_bf16(false, af, false, b0,
                                                     (short)0, acc0, false, false);
      acc1 = __builtin_amdgcn_wmma_f32_16x16x32_bf16(false, af, false, b1,
                                                     (short)0, acc1, false, false);
    }
  }
  const int na = n0 + l15;
  const int nb = n0 + 16 + l15;
  const float bb0 = bias[na];
  const float bb1 = bias[nb];
#pragma unroll
  for (int r = 0; r < 8; ++r) {
    const int m = m0 + r + 8 * half;
    float v0 = acc0[r] + bb0;
    float v1 = acc1[r] + bb1;
    if constexpr (RESID) {
      v0 += resid[(size_t)m * N + na];
      v1 += resid[(size_t)m * N + nb];
    }
    if constexpr (ACT == 1) {
      v0 = 0.5f * v0 * (1.0f + erff(v0 * 0.70710678118654752f));
      v1 = 0.5f * v1 * (1.0f + erff(v1 * 0.70710678118654752f));
    }
    if constexpr (OUTF) {
      Cf[(size_t)m * N + na] = v0;
      Cf[(size_t)m * N + nb] = v1;
    }
    if constexpr (OUTBF) {
      Cbf[(size_t)m * N + na] = f2bf(v0);
      Cbf[(size_t)m * N + nb] = f2bf(v1);
    }
  }
}

// ---------------------------------------------------------------------------
// LayerNorm over rows of length D_. One block per row.
// ---------------------------------------------------------------------------
__global__ void __launch_bounds__(256)
ln_kernel(const float* __restrict__ x, const float* __restrict__ g,
          const float* __restrict__ b, bf16* __restrict__ outb, float* __restrict__ outf) {
  __shared__ float sm[256];
  const int row = blockIdx.x;
  const float* xr = x + (size_t)row * D_;
  float s = 0.0f;
  for (int i = threadIdx.x; i < D_; i += 256) s += xr[i];
  sm[threadIdx.x] = s; __syncthreads();
  for (int st = 128; st > 0; st >>= 1) {
    if (threadIdx.x < st) sm[threadIdx.x] += sm[threadIdx.x + st];
    __syncthreads();
  }
  const float mean = sm[0] / (float)D_;
  __syncthreads();
  float v = 0.0f;
  for (int i = threadIdx.x; i < D_; i += 256) { float d = xr[i] - mean; v += d * d; }
  sm[threadIdx.x] = v; __syncthreads();
  for (int st = 128; st > 0; st >>= 1) {
    if (threadIdx.x < st) sm[threadIdx.x] += sm[threadIdx.x + st];
    __syncthreads();
  }
  const float inv = rsqrtf(sm[0] / (float)D_ + EPSLN);
  for (int i = threadIdx.x; i < D_; i += 256) {
    float val = (xr[i] - mean) * inv * g[i] + b[i];
    if (outb) outb[(size_t)row * D_ + i] = f2bf(val);
    if (outf) outf[(size_t)row * D_ + i] = val;
  }
}

// ---------------------------------------------------------------------------
// Causal attention, one block per (batch, head). Flash-style online softmax.
// K/V staged to LDS via async global->LDS copies when available (ASYNCcnt),
// else plain loads. q,k,v,out bf16 [MTOK, D_].
// ---------------------------------------------------------------------------
__global__ void __launch_bounds__(160)
attn_kernel(const bf16* __restrict__ q, const bf16* __restrict__ k,
            const bf16* __restrict__ v, bf16* __restrict__ out) {
  __shared__ bf16 ks[S3T][HD_];       // 19.2 KB
  __shared__ bf16 vs[S3T][HD_];       // 19.2 KB
  const int bh = blockIdx.x;
  const int b = bh / NH_, h = bh % NH_;
#if defined(USE_ASYNC_LDS)
  for (int idx = threadIdx.x; idx < S3T * (HD_ / 8); idx += blockDim.x) {
    int t = idx >> 3, c = (idx & 7) * 8;            // 8 bf16 = 16B per copy
    size_t goff = ((size_t)(b * S3T + t)) * D_ + h * HD_ + c;
    __builtin_amdgcn_global_load_async_to_lds_b128(
        (__attribute__((address_space(1))) v4i*)(k + goff),
        (__attribute__((address_space(3))) v4i*)&ks[t][c], 0, 0);
    __builtin_amdgcn_global_load_async_to_lds_b128(
        (__attribute__((address_space(1))) v4i*)(v + goff),
        (__attribute__((address_space(3))) v4i*)&vs[t][c], 0, 0);
  }
#if __has_builtin(__builtin_amdgcn_s_wait_asynccnt)
  __builtin_amdgcn_s_wait_asynccnt(0);
#else
  asm volatile("s_wait_asynccnt 0" ::: "memory");
#endif
#else
  for (int idx = threadIdx.x; idx < S3T * HD_; idx += blockDim.x) {
    int t = idx / HD_, d = idx % HD_;
    size_t goff = ((size_t)(b * S3T + t)) * D_ + h * HD_ + d;
    ks[t][d] = k[goff];
    vs[t][d] = v[goff];
  }
#endif
  __syncthreads();
  const int t = threadIdx.x;
  if (t < S3T) {
    const size_t qoff = ((size_t)(b * S3T + t)) * D_ + h * HD_;
    float qr[HD_];
#pragma unroll
    for (int d = 0; d < HD_; ++d) qr[d] = bf2f(q[qoff + d]) * 0.125f; // 1/sqrt(64)
    float m = -1e30f, l = 0.0f;
    float o[HD_];
#pragma unroll
    for (int d = 0; d < HD_; ++d) o[d] = 0.0f;
    for (int j = 0; j <= t; ++j) {
      float s = 0.0f;
#pragma unroll
      for (int d = 0; d < HD_; ++d) s += qr[d] * bf2f(ks[j][d]);
      float mn = fmaxf(m, s);
      float corr = __expf(m - mn);
      float p = __expf(s - mn);
      l = l * corr + p;
#pragma unroll
      for (int d = 0; d < HD_; ++d) o[d] = o[d] * corr + p * bf2f(vs[j][d]);
      m = mn;
    }
    const float inv = 1.0f / l;
    bf16* op = out + qoff;
#pragma unroll
    for (int d = 0; d < HD_; ++d) op[d] = f2bf(o[d] * inv);
  }
}

// ---------------------------------------------------------------------------
// State encoder conv1 (2x2, pad 1) + BN + ReLU + 2x2 maxpool, fused.
// in: [BT_,3,7,7] -> out: [BT_,128,4,4]
// ---------------------------------------------------------------------------
__global__ void se_conv1_kernel(const float* __restrict__ s, const float* __restrict__ W,
                                const float* __restrict__ cb, const float* __restrict__ bng,
                                const float* __restrict__ bnb, float* __restrict__ out) {
  int idx = blockIdx.x * blockDim.x + threadIdx.x;
  if (idx >= BT_ * 128 * 16) return;
  int pos = idx & 15, tmp = idx >> 4;
  int oc = tmp & 127, n = tmp >> 7;
  int py = pos >> 2, px = pos & 3;
  const float scale = bng[oc] * rsqrtf(1.0f + EPSLN);
  const float shift = bnb[oc];
  float mx = -1e30f;
  for (int dy = 0; dy < 2; ++dy)
    for (int dx = 0; dx < 2; ++dx) {
      int oy = 2 * py + dy, ox = 2 * px + dx;   // conv-out coords 0..7
      float acc = cb[oc];
      for (int c = 0; c < 3; ++c)
        for (int kh = 0; kh < 2; ++kh) {
          int iy = oy - 1 + kh; if (iy < 0 || iy > 6) continue;
          for (int kw = 0; kw < 2; ++kw) {
            int ix = ox - 1 + kw; if (ix < 0 || ix > 6) continue;
            acc += s[(size_t)n * 147 + c * 49 + iy * 7 + ix] *
                   W[((oc * 3 + c) * 2 + kh) * 2 + kw];
          }
        }
      float vv = fmaxf(acc * scale + shift, 0.0f);
      mx = fmaxf(mx, vv);
    }
  out[idx] = mx;
}

// ---------------------------------------------------------------------------
// State encoder conv2 (3x3, pad 1) + BN + ReLU + 2x2 maxpool, fused.
// in: [BT_,128,4,4] -> out: [BT_,128,2,2]
// ---------------------------------------------------------------------------
__global__ void se_conv2_kernel(const float* __restrict__ in, const float* __restrict__ W,
                                const float* __restrict__ cb, const float* __restrict__ bng,
                                const float* __restrict__ bnb, float* __restrict__ out) {
  int idx = blockIdx.x * blockDim.x + threadIdx.x;
  if (idx >= BT_ * 128 * 4) return;
  int pos = idx & 3, tmp = idx >> 2;
  int oc = tmp & 127, n = tmp >> 7;
  int py = pos >> 1, px = pos & 1;
  const float scale = bng[oc] * rsqrtf(1.0f + EPSLN);
  const float shift = bnb[oc];
  const float* wp = W + (size_t)oc * 128 * 9;
  float mx = -1e30f;
  for (int dy = 0; dy < 2; ++dy)
    for (int dx = 0; dx < 2; ++dx) {
      int oy = 2 * py + dy, ox = 2 * px + dx;   // 0..3
      float acc = cb[oc];
      for (int c = 0; c < 128; ++c) {
        const float* ip = in + ((size_t)n * 128 + c) * 16;
        for (int kh = 0; kh < 3; ++kh) {
          int iy = oy - 1 + kh; if (iy < 0 || iy > 3) continue;
          for (int kw = 0; kw < 3; ++kw) {
            int ix = ox - 1 + kw; if (ix < 0 || ix > 3) continue;
            acc += ip[iy * 4 + ix] * wp[c * 9 + kh * 3 + kw];
          }
        }
      }
      float vv = fmaxf(acc * scale + shift, 0.0f);
      mx = fmaxf(mx, vv);
    }
  out[idx] = mx;
}

// ---------------------------------------------------------------------------
// FiLM conv (3x3, pad 1) on 2x2 spatial; optional FiLM modulation; BN + ReLU.
// in: [N,128,2,2]; out: [N,OC,2,2]; gamma/beta: [B_,OC] (indexed by n / T_) or null.
// ---------------------------------------------------------------------------
__global__ void film_conv_kernel(const float* __restrict__ in, const float* __restrict__ W,
                                 const float* __restrict__ cb, const float* __restrict__ bng,
                                 const float* __restrict__ bnb, const float* __restrict__ gamma,
                                 const float* __restrict__ beta, float* __restrict__ out, int OC) {
  int idx = blockIdx.x * blockDim.x + threadIdx.x;
  if (idx >= BT_ * OC * 4) return;
  int pos = idx & 3, tmp = idx >> 2;
  int oc = tmp % OC, n = tmp / OC;
  int oy = pos >> 1, ox = pos & 1;
  const float* wp = W + (size_t)oc * 128 * 9;
  float acc = cb[oc];
  for (int c = 0; c < 128; ++c) {
    const float* ip = in + ((size_t)n * 128 + c) * 4;
    for (int kh = 0; kh < 3; ++kh) {
      int iy = oy - 1 + kh; if (iy < 0 || iy > 1) continue;
      for (int kw = 0; kw < 3; ++kw) {
        int ix = ox - 1 + kw; if (ix < 0 || ix > 1) continue;
        acc += ip[iy * 2 + ix] * wp[c * 9 + kh * 3 + kw];
      }
    }
  }
  if (gamma) {
    int b = n / T_;
    acc = acc * gamma[(size_t)b * OC + oc] + beta[(size_t)b * OC + oc];
  }
  float scale = bng[oc] * rsqrtf(1.0f + EPSLN);
  out[idx] = fmaxf(acc * scale + bnb[oc], 0.0f);
}

// relu(pool2) on [BT_,OC,2,2] -> [BT_,OC]  (inputs already >= 0)
__global__ void pool_flatten_kernel(const float* __restrict__ in, float* __restrict__ out, int OC) {
  int idx = blockIdx.x * blockDim.x + threadIdx.x;
  if (idx >= BT_ * OC) return;
  const float* p = in + (size_t)idx * 4;
  out[idx] = fmaxf(fmaxf(p[0], p[1]), fmaxf(p[2], p[3]));
}

// ---------------------------------------------------------------------------
// FiLM gamma/beta: out[b,oc] = instr[b,:] @ W[:,oc] + bias[oc]
// ---------------------------------------------------------------------------
__global__ void film_proj_kernel(const float* __restrict__ instr, const float* __restrict__ W,
                                 const float* __restrict__ bias, float* __restrict__ out, int OC) {
  int idx = blockIdx.x * blockDim.x + threadIdx.x;
  if (idx >= B_ * OC) return;
  int oc = idx % OC, b = idx / OC;
  const float* ir = instr + (size_t)b * D_;
  float s = bias[oc];
  for (int d = 0; d < D_; ++d) s += ir[d] * W[(size_t)d * OC + oc];
  out[idx] = s;
}

// ---------------------------------------------------------------------------
// GRU: embedding gather + one recurrent step.
// ---------------------------------------------------------------------------
__global__ void gru_gather_kernel(const int* __restrict__ insts, const float* __restrict__ word_emb,
                                  float* __restrict__ emb) {
  int idx = blockIdx.x * blockDim.x + threadIdx.x;     // B*IL*D
  if (idx >= B_ * IL_ * D_) return;
  int d = idx % D_, r = idx / D_;
  emb[idx] = word_emb[(size_t)insts[r] * D_ + d];
}

__global__ void gru_step_kernel(const float* __restrict__ emb, int t,
                                const float* __restrict__ Wih, const float* __restrict__ Whh,
                                const float* __restrict__ bih, const float* __restrict__ bhh,
                                const float* __restrict__ hprev, float* __restrict__ hnew) {
  int idx = blockIdx.x * blockDim.x + threadIdx.x;     // B*D
  if (idx >= B_ * D_) return;
  int j = idx % D_, b = idx / D_;
  const float* xt = emb + ((size_t)b * IL_ + t) * D_;
  const float* hp = hprev + (size_t)b * D_;
  float gir = bih[j], giz = bih[D_ + j], gin = bih[2 * D_ + j];
  float ghr = bhh[j], ghz = bhh[D_ + j], ghn = bhh[2 * D_ + j];
  const float* Wr = Wih + (size_t)j * D_;
  const float* Wz = Wih + (size_t)(D_ + j) * D_;
  const float* Wn = Wih + (size_t)(2 * D_ + j) * D_;
  const float* Ur = Whh + (size_t)j * D_;
  const float* Uz = Whh + (size_t)(D_ + j) * D_;
  const float* Un = Whh + (size_t)(2 * D_ + j) * D_;
  for (int kk = 0; kk < D_; ++kk) {
    float xv = xt[kk], hv = hp[kk];
    gir += xv * Wr[kk]; giz += xv * Wz[kk]; gin += xv * Wn[kk];
    ghr += hv * Ur[kk]; ghz += hv * Uz[kk]; ghn += hv * Un[kk];
  }
  float r = sigmoidf_(gir + ghr);
  float z = sigmoidf_(giz + ghz);
  float nn = tanhf(gin + r * ghn);
  hnew[idx] = (1.0f - z) * nn + z * hp[j];
}

// ---------------------------------------------------------------------------
// Token interleave (rtg, state, action) + positional embeddings -> x [MTOK, D_]
// ---------------------------------------------------------------------------
__global__ void build_tokens_kernel(const float* __restrict__ rtgs, const int* __restrict__ actions,
                                    const int* __restrict__ timesteps, const float* __restrict__ state_emb,
                                    const float* __restrict__ ret_W, const float* __restrict__ ret_b,
                                    const float* __restrict__ act_emb_W, const float* __restrict__ pos_emb,
                                    const float* __restrict__ gpos_emb, float* __restrict__ x) {
  int idx = blockIdx.x * blockDim.x + threadIdx.x;     // MTOK * D_
  if (idx >= MTOK * D_) return;
  int d = idx % D_;
  int row = idx / D_;
  int b = row / S3T, s = row % S3T;
  int t = s / 3, kind = s % 3;
  float tok;
  if (kind == 0)      tok = tanhf(rtgs[b * T_ + t] * ret_W[d] + ret_b[d]);
  else if (kind == 1) tok = state_emb[((size_t)(b * T_ + t)) * D_ + d];
  else                tok = tanhf(act_emb_W[(size_t)actions[b * T_ + t] * D_ + d]);
  int ts = timesteps[b];
  float pos = gpos_emb[(size_t)ts * D_ + d] + pos_emb[(size_t)s * D_ + d];
  x[idx] = tok + pos;
}

// ---------------------------------------------------------------------------
// Head: logits from state tokens only. out [B_,T_,V_]
// ---------------------------------------------------------------------------
__global__ void head_kernel(const float* __restrict__ xf, const float* __restrict__ head_W,
                            float* __restrict__ out) {
  int idx = blockIdx.x * blockDim.x + threadIdx.x;     // BT_ * V_
  if (idx >= BT_ * V_) return;
  int v = idx % V_, r = idx / V_;
  int b = r / T_, t = r % T_;
  const float* xr = xf + ((size_t)(b * S3T + 1 + 3 * t)) * D_;
  float s = 0.0f;
  for (int kk = 0; kk < D_; ++kk) s += xr[kk] * head_W[kk * V_ + v];
  out[idx] = s;
}

// ---------------------------------------------------------------------------
// Host side
// ---------------------------------------------------------------------------
static inline dim3 g1d(size_t n, int blk) { return dim3((unsigned)((n + blk - 1) / blk)); }

extern "C" void kernel_launch(void* const* d_in, const int* in_sizes, int n_in,
                              void* d_out, int out_size, void* d_ws, size_t ws_size,
                              hipStream_t stream) {
  (void)in_sizes; (void)n_in; (void)out_size; (void)ws_size;
  const float* states     = (const float*)d_in[0];
  const int*   actions    = (const int*)  d_in[1];
  const float* rtgs       = (const float*)d_in[3];
  const int*   timesteps  = (const int*)  d_in[4];
  const int*   insts      = (const int*)  d_in[5];
  const float* se_c1_W = (const float*)d_in[6],  *se_c1_b = (const float*)d_in[7];
  const float* se_bn1_g= (const float*)d_in[8],  *se_bn1_b= (const float*)d_in[9];
  const float* se_c2_W = (const float*)d_in[10], *se_c2_b = (const float*)d_in[11];
  const float* se_bn2_g= (const float*)d_in[12], *se_bn2_b= (const float*)d_in[13];
  const float* word_emb= (const float*)d_in[14];
  const float* gru_Wih = (const float*)d_in[15], *gru_Whh = (const float*)d_in[16];
  const float* gru_bih = (const float*)d_in[17], *gru_bhh = (const float*)d_in[18];
  const float* f0_c1_W = (const float*)d_in[19], *f0_c1_b = (const float*)d_in[20];
  const float* f0_bn1_g= (const float*)d_in[21], *f0_bn1_b= (const float*)d_in[22];
  const float* f0_c2_W = (const float*)d_in[23], *f0_c2_b = (const float*)d_in[24];
  const float* f0_bn2_g= (const float*)d_in[25], *f0_bn2_b= (const float*)d_in[26];
  const float* f0_wW   = (const float*)d_in[27], *f0_wb   = (const float*)d_in[28];
  const float* f0_bW   = (const float*)d_in[29], *f0_bb   = (const float*)d_in[30];
  const float* f1_c1_W = (const float*)d_in[31], *f1_c1_b = (const float*)d_in[32];
  const float* f1_bn1_g= (const float*)d_in[33], *f1_bn1_b= (const float*)d_in[34];
  const float* f1_c2_W = (const float*)d_in[35], *f1_c2_b = (const float*)d_in[36];
  const float* f1_bn2_g= (const float*)d_in[37], *f1_bn2_b= (const float*)d_in[38];
  const float* f1_wW   = (const float*)d_in[39], *f1_wb   = (const float*)d_in[40];
  const float* f1_bW   = (const float*)d_in[41], *f1_bb   = (const float*)d_in[42];
  const float* ret_W   = (const float*)d_in[43], *ret_b   = (const float*)d_in[44];
  const float* act_emb_W = (const float*)d_in[45];
  const float* pos_emb = (const float*)d_in[46], *gpos_emb = (const float*)d_in[47];
  const float* ln1_g = (const float*)d_in[48], *ln1_b = (const float*)d_in[49];
  const float* ln2_g = (const float*)d_in[50], *ln2_b = (const float*)d_in[51];
  const float* Wq = (const float*)d_in[52], *Wk = (const float*)d_in[53];
  const float* Wv = (const float*)d_in[54], *Wo = (const float*)d_in[55];
  const float* bq = (const float*)d_in[56], *bk = (const float*)d_in[57];
  const float* bv = (const float*)d_in[58], *bo = (const float*)d_in[59];
  const float* mlp_W1 = (const float*)d_in[60], *mlp_b1 = (const float*)d_in[61];
  const float* mlp_W2 = (const float*)d_in[62], *mlp_b2 = (const float*)d_in[63];
  const float* lnf_g = (const float*)d_in[64], *lnf_b = (const float*)d_in[65];
  const float* head_W = (const float*)d_in[66];
  float* out = (float*)d_out;

  // ---- workspace layout (~350 MB) ----
  char* ws = (char*)d_ws;
  size_t off = 0;
  auto alloc = [&](size_t bytes) -> char* {
    char* p = ws + off;
    off = (off + bytes + 255) & ~((size_t)255);
    return p;
  };
  const size_t WPL = 3 * 2359296;                        // bf16 elems per layer
  bf16*  wbf      = (bf16*) alloc((size_t)NL_ * WPL * 2);
  float* x        = (float*)alloc((size_t)MTOK * D_ * 4);
  bf16*  ln_bf    = (bf16*) alloc((size_t)MTOK * D_ * 2);
  bf16*  att_bf   = (bf16*) alloc((size_t)MTOK * D_ * 2);
  char*  arena    = alloc((size_t)MTOK * 4 * D_ * 2);    // 117,964,800 B, time-multiplexed
  float* state_emb= (float*)alloc((size_t)BT_ * D_ * 4);
  float* gru_emb  = (float*)alloc((size_t)B_ * IL_ * D_ * 4);
  float* hA       = (float*)alloc((size_t)B_ * D_ * 4);
  float* hB       = (float*)alloc((size_t)B_ * D_ * 4);
  float* gam0     = (float*)alloc((size_t)B_ * 128 * 4);
  float* bet0     = (float*)alloc((size_t)B_ * 128 * 4);
  float* gam1     = (float*)alloc((size_t)B_ * D_ * 4);
  float* bet1     = (float*)alloc((size_t)B_ * D_ * 4);

  // arena sub-views (time-multiplexed: CNN temps -> q/k/v -> MLP hidden -> lnf)
  float* pool1    = (float*)(arena);                       // [BT_,128,4,4]
  float* pool2    = (float*)(arena + 52428800);            // [BT_,128,2,2]
  float* filmA_t  = (float*)(arena);                       // reuses pool1 space
  float* film0_o  = (float*)(arena + 13107200);
  float* filmB_t  = (float*)(arena + 26214400);
  float* film1_o  = (float*)(arena + 39321600);            // [BT_,768,2,2]
  bf16*  q_bf     = (bf16*)(arena);
  bf16*  k_bf     = (bf16*)(arena + 29491200);
  bf16*  v_bf     = (bf16*)(arena + 58982400);
  bf16*  h1g_bf   = (bf16*)(arena);                        // [MTOK, 4D] bf16
  float* lnf_f    = (float*)(arena);                       // [MTOK, D] f32

  auto cvtB = [&](const float* src, bf16* dst, int K, int N) {
    convert_swizzle_b_kernel<<<g1d((size_t)K * N, 256), 256, 0, stream>>>(src, dst, K, N);
  };

  // ---- 1. convert + swizzle transformer weights to bf16 fragment layout ----
  for (int l = 0; l < NL_; ++l) {
    bf16* base = wbf + (size_t)l * WPL;
    cvtB(Wq + (size_t)l * D_ * D_, base + 0,       D_, D_);
    cvtB(Wk + (size_t)l * D_ * D_, base + 589824,  D_, D_);
    cvtB(Wv + (size_t)l * D_ * D_, base + 1179648, D_, D_);
    cvtB(Wo + (size_t)l * D_ * D_, base + 1769472, D_, D_);
    cvtB(mlp_W1 + (size_t)l * D_ * 4 * D_, base + 2359296, D_, 4 * D_);
    cvtB(mlp_W2 + (size_t)l * 4 * D_ * D_, base + 4718592, 4 * D_, D_);
  }

  // ---- 2. state-encoder CNN ----
  se_conv1_kernel<<<g1d((size_t)BT_ * 128 * 16, 256), 256, 0, stream>>>(
      states, se_c1_W, se_c1_b, se_bn1_g, se_bn1_b, pool1);
  se_conv2_kernel<<<g1d((size_t)BT_ * 128 * 4, 256), 256, 0, stream>>>(
      pool1, se_c2_W, se_c2_b, se_bn2_g, se_bn2_b, pool2);

  // ---- 3. instruction GRU ----
  gru_gather_kernel<<<g1d((size_t)B_ * IL_ * D_, 256), 256, 0, stream>>>(insts, word_emb, gru_emb);
  zero_kernel<<<g1d((size_t)B_ * D_, 256), 256, 0, stream>>>(hA, B_ * D_);
  float* hc = hA; float* hn = hB;
  for (int t = 0; t < IL_; ++t) {
    gru_step_kernel<<<g1d((size_t)B_ * D_, 256), 256, 0, stream>>>(
        gru_emb, t, gru_Wih, gru_Whh, gru_bih, gru_bhh, hc, hn);
    float* tmp = hc; hc = hn; hn = tmp;
  }
  const float* instr = hc;   // final hidden [B_, D_]

  // ---- 4. FiLM modulation coefficients ----
  film_proj_kernel<<<g1d((size_t)B_ * 128, 256), 256, 0, stream>>>(instr, f0_wW, f0_wb, gam0, 128);
  film_proj_kernel<<<g1d((size_t)B_ * 128, 256), 256, 0, stream>>>(instr, f0_bW, f0_bb, bet0, 128);
  film_proj_kernel<<<g1d((size_t)B_ * D_, 256), 256, 0, stream>>>(instr, f1_wW, f1_wb, gam1, D_);
  film_proj_kernel<<<g1d((size_t)B_ * D_, 256), 256, 0, stream>>>(instr, f1_bW, f1_bb, bet1, D_);

  // ---- 5. FiLM blocks ----
  film_conv_kernel<<<g1d((size_t)BT_ * 128 * 4, 256), 256, 0, stream>>>(
      pool2, f0_c1_W, f0_c1_b, f0_bn1_g, f0_bn1_b, nullptr, nullptr, filmA_t, 128);
  film_conv_kernel<<<g1d((size_t)BT_ * 128 * 4, 256), 256, 0, stream>>>(
      filmA_t, f0_c2_W, f0_c2_b, f0_bn2_g, f0_bn2_b, gam0, bet0, film0_o, 128);
  film_conv_kernel<<<g1d((size_t)BT_ * 128 * 4, 256), 256, 0, stream>>>(
      film0_o, f1_c1_W, f1_c1_b, f1_bn1_g, f1_bn1_b, nullptr, nullptr, filmB_t, 128);
  film_conv_kernel<<<g1d((size_t)BT_ * D_ * 4, 256), 256, 0, stream>>>(
      filmB_t, f1_c2_W, f1_c2_b, f1_bn2_g, f1_bn2_b, gam1, bet1, film1_o, D_);
  pool_flatten_kernel<<<g1d((size_t)BT_ * D_, 256), 256, 0, stream>>>(film1_o, state_emb, D_);

  // ---- 6. token embedding + positions ----
  build_tokens_kernel<<<g1d((size_t)MTOK * D_, 256), 256, 0, stream>>>(
      rtgs, actions, timesteps, state_emb, ret_W, ret_b, act_emb_W, pos_emb, gpos_emb, x);

  // ---- 7. transformer layers ----
  for (int l = 0; l < NL_; ++l) {
    bf16* base = wbf + (size_t)l * WPL;
    const bf16 *wq = base, *wk = base + 589824, *wv = base + 1179648, *wo = base + 1769472;
    const bf16 *w1 = base + 2359296, *w2 = base + 4718592;
    dim3 gQ(MTOK / 64, D_ / 64), gM1(MTOK / 64, 4 * D_ / 64);

    ln_kernel<<<MTOK, 256, 0, stream>>>(x, ln1_g + l * D_, ln1_b + l * D_, ln_bf, nullptr);
    wmma_gemm_bf16<0, false, false, true><<<gQ, 256, 0, stream>>>(
        ln_bf, wq, bq + l * D_, nullptr, nullptr, q_bf, MTOK, D_, D_);
    wmma_gemm_bf16<0, false, false, true><<<gQ, 256, 0, stream>>>(
        ln_bf, wk, bk + l * D_, nullptr, nullptr, k_bf, MTOK, D_, D_);
    wmma_gemm_bf16<0, false, false, true><<<gQ, 256, 0, stream>>>(
        ln_bf, wv, bv + l * D_, nullptr, nullptr, v_bf, MTOK, D_, D_);
    attn_kernel<<<B_ * NH_, 160, 0, stream>>>(q_bf, k_bf, v_bf, att_bf);
    wmma_gemm_bf16<0, true, true, false><<<gQ, 256, 0, stream>>>(
        att_bf, wo, bo + l * D_, x, x, nullptr, MTOK, D_, D_);

    ln_kernel<<<MTOK, 256, 0, stream>>>(x, ln2_g + l * D_, ln2_b + l * D_, ln_bf, nullptr);
    wmma_gemm_bf16<1, false, false, true><<<gM1, 256, 0, stream>>>(
        ln_bf, w1, mlp_b1 + (size_t)l * 4 * D_, nullptr, nullptr, h1g_bf, MTOK, 4 * D_, D_);
    wmma_gemm_bf16<0, true, true, false><<<gQ, 256, 0, stream>>>(
        h1g_bf, w2, mlp_b2 + l * D_, x, x, nullptr, MTOK, D_, 4 * D_);
  }

  // ---- 8. final LN + head ----
  ln_kernel<<<MTOK, 256, 0, stream>>>(x, lnf_g, lnf_b, nullptr, lnf_f);
  head_kernel<<<g1d((size_t)BT_ * V_, 256), 256, 0, stream>>>(lnf_f, head_W, out);
}
